// FourierAttention2d_44169443672697
// MI455X (gfx1250) — matrix operand order repaired and checked by reading, not compiled, not run on hardware
//
#include <hip/hip_runtime.h>
#include <hip/hip_bf16.h>

typedef __attribute__((ext_vector_type(16))) _Float16 v16h;
typedef __attribute__((ext_vector_type(8)))  _Float16 v8h;
typedef __attribute__((ext_vector_type(8)))  float    v8f;

#define B_   4
#define N_   16384
#define CIN_ 64
#define H_   8
#define D_   64
#define HD_  512
#define EPS_ 1e-5f

// ---------------- WMMA helpers (CDNA5 gfx1250, wave32) ----------------

__device__ __forceinline__ v8f wmma_f16(v16h a, v16h b, v8f c) {
  // D(f32, 16x16) = A(16x32 f16) * B(32x16 f16) + C
  return __builtin_amdgcn_wmma_f32_16x16x32_f16(false, a, false, b, (short)0, c,
                                                false, false);
}

// A fragment (16x32 f16) from row-major f16 [row][k], ld in halfs (mult of 8, rows 16B aligned)
// ISA layout: lane r (0..15), group g = lane>>4: halves 0..7 -> K = g*8+i, halves 8..15 -> K = 16+g*8+i
__device__ __forceinline__ v16h load_a_rm_h(const _Float16* p0, int ld) {
  const int lane = threadIdx.x & 31;
  const int g = lane >> 4, r = lane & 15;
  const _Float16* p = p0 + r * ld + g * 8;
  v8h lo = *(const v8h*)(p);
  v8h hi = *(const v8h*)(p + 16);
  v16h a;
#pragma unroll
  for (int i = 0; i < 8; ++i) { a[i] = lo[i]; a[i + 8] = hi[i]; }
  return a;
}

// A fragment from row-major f32 (converted to f16)
__device__ __forceinline__ v16h load_a_rm_f32(const float* p0, int ld) {
  const int lane = threadIdx.x & 31;
  const int g = lane >> 4, r = lane & 15;
  const float* p = p0 + r * ld + g * 8;
  v16h a;
#pragma unroll
  for (int i = 0; i < 8; ++i) {
    a[i]     = (_Float16)p[i];
    a[i + 8] = (_Float16)p[16 + i];
  }
  return a;
}

// B fragment (32x16 f16) from N-major ("transposed") f16 [n][k], ld in halfs
// ISA layout: lane c = lane&15 -> column N=c; halves 0..7 -> K = g*8+i, halves 8..15 -> K = 16+g*8+i
__device__ __forceinline__ v16h load_b_nm_h(const _Float16* p0, int ld) {
  const int lane = threadIdx.x & 31;
  const int g = lane >> 4, c = lane & 15;
  const _Float16* p = p0 + c * ld + g * 8;
  v8h lo = *(const v8h*)(p);
  v8h hi = *(const v8h*)(p + 16);
  v16h b;
#pragma unroll
  for (int i = 0; i < 8; ++i) { b[i] = lo[i]; b[i + 8] = hi[i]; }
  return b;
}

// sum across the 16 lanes of a half-wave group (wave32; xor masks < 16 stay in group)
__device__ __forceinline__ float hred16(float x) {
#pragma unroll
  for (int m = 8; m >= 1; m >>= 1) x += __shfl_xor(x, m, 32);
  return x;
}

// Async-copy 32 rows x 64 f32 (8KB, globally contiguous) into LDS with row stride
// 68 floats (272B, 16B-aligned). Uses gfx1250 GLOBAL_LOAD_ASYNC_TO_LDS_B128 (ASYNCcnt).
__device__ __forceinline__ void async_stage_rows32(const float* gsrc, float* ldst) {
  const int lane = threadIdx.x & 31;
#pragma unroll
  for (int s = 0; s < 16; ++s) {
    int c = s * 32 + lane;            // 16B-chunk index, 0..511
    int row = c >> 4, col = c & 15;   // 16 chunks per 256B row
    unsigned lds = (unsigned)(size_t)(ldst + row * 68) + col * 16;
    const char* g = (const char*)gsrc + c * 16;
    asm volatile("global_load_async_to_lds_b128 %0, %1, off"
                 :: "v"(lds), "v"(g) : "memory");
  }
  asm volatile("s_wait_asynccnt 0" ::: "memory");
}

// ---------------- kernel 0: weight transpose + f16 convert ----------------
// WqT/WkT/WvT: [512][64] f16 (row = output col, entries = cin)
// WoT:        [64][512] f16 (row = out channel e, entries = h*64+d)
__global__ __launch_bounds__(256) void prep_kernel(
    const float* __restrict__ Wq, const float* __restrict__ Wk,
    const float* __restrict__ Wv, const float* __restrict__ Wo,
    _Float16* __restrict__ WqT, _Float16* __restrict__ WkT,
    _Float16* __restrict__ WvT, _Float16* __restrict__ WoT) {
  int i = blockIdx.x * 256 + threadIdx.x;
  if (i < 32768) {
    int o = i >> 6, c = i & 63;
    WqT[o * 64 + c] = (_Float16)Wq[c * 512 + o];
    WkT[o * 64 + c] = (_Float16)Wk[c * 512 + o];
    WvT[o * 64 + c] = (_Float16)Wv[c * 512 + o];
    int e = i >> 9, op = i & 511;
    WoT[e * 512 + op] = (_Float16)Wo[op * 64 + e];
  }
}

// ---------------- kernel 1: fused K/V projection + LN(K) + kv partial ----------------
// grid: B*H*8 blocks of 128 threads (4 waves).
// Per iteration the block processes 128 rows: each wave projects+LNs its own 32-row
// strip, all strips are transposed into shared Kt/Vt [64][128], then wave w computes
// the d-strip [16w,16w+16) of kv (K-dim = 128). kv accum = 32 VGPRs/wave.
// part: [B][H][8 tiles][64][64] f32 (deterministic fixed-order reduce in mh_kernel)
__global__ __launch_bounds__(128) void kv_kernel(
    const float* __restrict__ v, const _Float16* __restrict__ WkT,
    const _Float16* __restrict__ WvT, const float* __restrict__ Wk_b,
    const float* __restrict__ Wv_b, const float* __restrict__ lnk_g,
    const float* __restrict__ lnk_b, float* __restrict__ part) {
  // union: [v staging: 4 waves x 32x68 f32 = 34816B] / [Kt+Vt: 2 x 64x136 f16 = 34816B]
  __shared__ alignas(16) unsigned char smem[34816];
  const int lane = threadIdx.x & 31;
  const int wv = threadIdx.x >> 5;
  const int g = lane >> 4, cl = lane & 15;

  const int tile = blockIdx.x & 7;
  const int h = (blockIdx.x >> 3) & 7;
  const int b = blockIdx.x >> 6;

  float*    vsm = (float*)smem + wv * (32 * 68);
  _Float16* Kt  = (_Float16*)smem;                 // [64][136]
  _Float16* Vt  = (_Float16*)(smem + 17408);       // [64][136]

  // per-lane column params (column = h*64 + nt*16 + cl)
  float bk[4], bv4[4], gk[4], bt[4];
#pragma unroll
  for (int nt = 0; nt < 4; ++nt) {
    int col = h * 64 + nt * 16 + cl;
    bk[nt] = Wk_b[col]; bv4[nt] = Wv_b[col];
    gk[nt] = lnk_g[col]; bt[nt] = lnk_b[col];
  }

  v8f kvacc[4];
#pragma unroll
  for (int et = 0; et < 4; ++et) { v8f z = {}; kvacc[et] = z; }

  const int chunk = N_ / 8;                 // 2048 rows per block
  const int iters = chunk / 128;            // 16 iterations of 128 rows
  const int base = b * N_ + tile * chunk;

  for (int it = 0; it < iters; ++it) {
    const int r0 = base + it * 128 + wv * 32;

    // stage this wave's 32 rows of v via async-to-LDS (f32)
    async_stage_rows32(v + (size_t)r0 * CIN_, vsm);

    // A fragments (2 row-tiles x 2 k-steps over cin=64), f32->f16 on the fly
    v16h af[2][2];
#pragma unroll
    for (int mt = 0; mt < 2; ++mt)
#pragma unroll
      for (int ks = 0; ks < 2; ++ks)
        af[mt][ks] = load_a_rm_f32(vsm + mt * 16 * 68 + ks * 32, 68);
    __syncthreads();   // v staging region is now dead; Kt/Vt may overwrite it

    // K and V projections
    v8f ck[2][4], cv[2][4];
#pragma unroll
    for (int mt = 0; mt < 2; ++mt)
#pragma unroll
      for (int nt = 0; nt < 4; ++nt) { v8f z = {}; ck[mt][nt] = z; cv[mt][nt] = z; }
#pragma unroll
    for (int nt = 0; nt < 4; ++nt)
#pragma unroll
      for (int ks = 0; ks < 2; ++ks) {
        v16h bkf = load_b_nm_h(WkT + (h * 64 + nt * 16) * 64 + ks * 32, 64);
        v16h bvf = load_b_nm_h(WvT + (h * 64 + nt * 16) * 64 + ks * 32, 64);
#pragma unroll
        for (int mt = 0; mt < 2; ++mt) {
          ck[mt][nt] = wmma_f16(af[mt][ks], bkf, ck[mt][nt]);
          cv[mt][nt] = wmma_f16(af[mt][ks], bvf, cv[mt][nt]);
        }
      }

    // bias + LayerNorm(K rows), bias on V
#pragma unroll
    for (int mt = 0; mt < 2; ++mt) {
#pragma unroll
      for (int nt = 0; nt < 4; ++nt)
#pragma unroll
        for (int j = 0; j < 8; ++j) {
          ck[mt][nt][j] += bk[nt];
          cv[mt][nt][j] += bv4[nt];
        }
#pragma unroll
      for (int j = 0; j < 8; ++j) {
        float s = 0.f, q = 0.f;
#pragma unroll
        for (int nt = 0; nt < 4; ++nt) { float x = ck[mt][nt][j]; s += x; q += x * x; }
        s = hred16(s); q = hred16(q);
        float mu = s * (1.f / 64.f);
        float var = q * (1.f / 64.f) - mu * mu;
        float rs = rsqrtf(var + EPS_);
#pragma unroll
        for (int nt = 0; nt < 4; ++nt) {
          float x = (ck[mt][nt][j] - mu) * rs;
          ck[mt][nt][j] = x * gk[nt] + bt[nt];
        }
      }
    }

    // transpose all waves' K,V strips into shared Kt/Vt[d][m], m = 0..127
#pragma unroll
    for (int mt = 0; mt < 2; ++mt)
#pragma unroll
      for (int nt = 0; nt < 4; ++nt)
#pragma unroll
        for (int j = 0; j < 8; ++j) {
          int m = wv * 32 + mt * 16 + g * 8 + j;
          int d = nt * 16 + cl;
          Kt[d * 136 + m] = (_Float16)ck[mt][nt][j];
          Vt[d * 136 + m] = (_Float16)cv[mt][nt][j];
        }
    __syncthreads();

    // kv(d-strip wv) += K^T * V over 128 rows (4 k-steps)
#pragma unroll
    for (int ks = 0; ks < 4; ++ks) {
      v16h ak = load_a_rm_h(Kt + (wv * 16) * 136 + ks * 32, 136);
#pragma unroll
      for (int et = 0; et < 4; ++et) {
        v16h bvf2 = load_b_nm_h(Vt + (et * 16) * 136 + ks * 32, 136);
        kvacc[et] = wmma_f16(ak, bvf2, kvacc[et]);
      }
    }
    __syncthreads();   // Kt/Vt reads done before next iteration's staging
  }

  // each wave owns disjoint kv rows -> direct store, fully deterministic
  float* outp = part + ((size_t)((b * 8 + h) * 8 + tile)) * 4096;
#pragma unroll
  for (int et = 0; et < 4; ++et)
#pragma unroll
    for (int j = 0; j < 8; ++j) {
      int d = wv * 16 + g * 8 + j, e = et * 16 + cl;
      outp[d * 64 + e] = kvacc[et][j];
    }
}

// ---------------- kernel 2: M_h = (kv_h @ Wo_h) / n, stored transposed f16 ----------------
// grid: B*H blocks of 32 threads. MT: [B][H][e=64][d=64] f16
__global__ __launch_bounds__(32) void mh_kernel(
    const float* __restrict__ part, const _Float16* __restrict__ WoT,
    _Float16* __restrict__ MT) {
  __shared__ alignas(16) float kvsm[64 * 64];
  const int lane = threadIdx.x & 31;
  const int g = lane >> 4, cl = lane & 15;
  const int h = blockIdx.x & 7, b = blockIdx.x >> 3;

  const float* p0 = part + (size_t)(b * 8 + h) * 8 * 4096;
  for (int i = lane; i < 4096; i += 32) {
    float s = 0.f;
#pragma unroll
    for (int t = 0; t < 8; ++t) s += p0[t * 4096 + i];
    kvsm[i] = s;
  }
  __syncthreads();

  v8f c[4][4];
#pragma unroll
  for (int mt = 0; mt < 4; ++mt)
#pragma unroll
    for (int nt = 0; nt < 4; ++nt) { v8f z = {}; c[mt][nt] = z; }

#pragma unroll
  for (int ks = 0; ks < 2; ++ks) {
    v16h a[4], bb[4];
#pragma unroll
    for (int mt = 0; mt < 4; ++mt)
      a[mt] = load_a_rm_f32(kvsm + mt * 16 * 64 + ks * 32, 64);
#pragma unroll
    for (int nt = 0; nt < 4; ++nt)
      bb[nt] = load_b_nm_h(WoT + (nt * 16) * 512 + h * 64 + ks * 32, 512);
#pragma unroll
    for (int mt = 0; mt < 4; ++mt)
#pragma unroll
      for (int nt = 0; nt < 4; ++nt)
        c[mt][nt] = wmma_f16(a[mt], bb[nt], c[mt][nt]);
  }

  _Float16* outp = MT + (size_t)(b * 8 + h) * 4096;
#pragma unroll
  for (int mt = 0; mt < 4; ++mt)
#pragma unroll
    for (int nt = 0; nt < 4; ++nt)
#pragma unroll
      for (int j = 0; j < 8; ++j) {
        int d = mt * 16 + g * 8 + j, e = nt * 16 + cl;
        outp[e * 64 + d] = (_Float16)(c[mt][nt][j] * (1.f / 16384.f));
      }
}

// ---------------- kernel 3: fused Q projection + LN + out = sum_h Q_h M_h ----------------
// grid: B*128 blocks of 128 threads (4 waves x 32 rows). Per-wave LDS regions only;
// DS ops are in-order within a wave, so no barriers are needed.
__global__ __launch_bounds__(128) void out_kernel(
    const float* __restrict__ v, const _Float16* __restrict__ WqT,
    const float* __restrict__ Wq_b, const float* __restrict__ lnq_g,
    const float* __restrict__ lnq_b, const _Float16* __restrict__ MT,
    const float* __restrict__ Wo_b, float* __restrict__ out) {
  __shared__ alignas(16) unsigned char smem[34816 + 18432];
  const int lane = threadIdx.x & 31;
  const int wv = threadIdx.x >> 5;
  const int g = lane >> 4, cl = lane & 15;
  const int b = blockIdx.x >> 7;
  const int tile = blockIdx.x & 127;
  const int r0 = b * N_ + tile * 128 + wv * 32;

  float*    vsm = (float*)smem + wv * (32 * 68);            // 32 x 68 f32
  _Float16* Qsm = (_Float16*)(smem + 34816) + wv * 2304;    // 32 x 72 f16

  // stage this wave's 32 rows of v via async-to-LDS
  async_stage_rows32(v + (size_t)r0 * CIN_, vsm);

  v16h af[2][2];
#pragma unroll
  for (int mt = 0; mt < 2; ++mt)
#pragma unroll
    for (int ks = 0; ks < 2; ++ks)
      af[mt][ks] = load_a_rm_f32(vsm + mt * 16 * 68 + ks * 32, 68);

  v8f co[2][4];
#pragma unroll
  for (int mt = 0; mt < 2; ++mt)
#pragma unroll
    for (int nt = 0; nt < 4; ++nt) { v8f z = {}; co[mt][nt] = z; }

  for (int h = 0; h < 8; ++h) {
    float bq[4], gq[4], btq[4];
#pragma unroll
    for (int nt = 0; nt < 4; ++nt) {
      int col = h * 64 + nt * 16 + cl;
      bq[nt] = Wq_b[col]; gq[nt] = lnq_g[col]; btq[nt] = lnq_b[col];
    }

    // Q projection for head h
    v8f cq[2][4];
#pragma unroll
    for (int mt = 0; mt < 2; ++mt)
#pragma unroll
      for (int nt = 0; nt < 4; ++nt) { v8f z = {}; cq[mt][nt] = z; }
#pragma unroll
    for (int nt = 0; nt < 4; ++nt)
#pragma unroll
      for (int ks = 0; ks < 2; ++ks) {
        v16h bf = load_b_nm_h(WqT + (h * 64 + nt * 16) * 64 + ks * 32, 64);
#pragma unroll
        for (int mt = 0; mt < 2; ++mt)
          cq[mt][nt] = wmma_f16(af[mt][ks], bf, cq[mt][nt]);
      }

    // bias + LayerNorm
#pragma unroll
    for (int mt = 0; mt < 2; ++mt) {
#pragma unroll
      for (int nt = 0; nt < 4; ++nt)
#pragma unroll
        for (int j = 0; j < 8; ++j) cq[mt][nt][j] += bq[nt];
#pragma unroll
      for (int j = 0; j < 8; ++j) {
        float s = 0.f, q = 0.f;
#pragma unroll
        for (int nt = 0; nt < 4; ++nt) { float x = cq[mt][nt][j]; s += x; q += x * x; }
        s = hred16(s); q = hred16(q);
        float mu = s * (1.f / 64.f);
        float var = q * (1.f / 64.f) - mu * mu;
        float rs = rsqrtf(var + EPS_);
#pragma unroll
        for (int nt = 0; nt < 4; ++nt) {
          float x = (cq[mt][nt][j] - mu) * rs;
          cq[mt][nt][j] = x * gq[nt] + btq[nt];
        }
      }
    }

    // Q rows -> per-wave LDS (row-major f16) for A fragments (same-wave DS is in-order)
#pragma unroll
    for (int mt = 0; mt < 2; ++mt)
#pragma unroll
      for (int nt = 0; nt < 4; ++nt)
#pragma unroll
        for (int j = 0; j < 8; ++j) {
          int m = mt * 16 + g * 8 + j;
          int col = nt * 16 + cl;
          Qsm[m * 72 + col] = (_Float16)cq[mt][nt][j];
        }

    // out += Q_h @ M_h
    const _Float16* mh = MT + (size_t)(b * 8 + h) * 4096;
    v16h aq[2][2];
#pragma unroll
    for (int mt = 0; mt < 2; ++mt)
#pragma unroll
      for (int ks = 0; ks < 2; ++ks)
        aq[mt][ks] = load_a_rm_h(Qsm + mt * 16 * 72 + ks * 32, 72);
#pragma unroll
    for (int nt = 0; nt < 4; ++nt)
#pragma unroll
      for (int ks = 0; ks < 2; ++ks) {
        v16h bm = load_b_nm_h(mh + (nt * 16) * 64 + ks * 32, 64);
#pragma unroll
        for (int mt = 0; mt < 2; ++mt)
          co[mt][nt] = wmma_f16(aq[mt][ks], bm, co[mt][nt]);
      }
  }

  // epilogue: + Wo_b / n, non-temporal stores
  float* o0 = out + (size_t)r0 * 64;
#pragma unroll
  for (int mt = 0; mt < 2; ++mt)
#pragma unroll
    for (int nt = 0; nt < 4; ++nt)
#pragma unroll
      for (int j = 0; j < 8; ++j) {
        int m = mt * 16 + g * 8 + j, e = nt * 16 + cl;
        float val = co[mt][nt][j] + Wo_b[e] * (1.f / 16384.f);
        __builtin_nontemporal_store(val, &o0[m * 64 + e]);
      }
}

// ---------------- host launch ----------------
extern "C" void kernel_launch(void* const* d_in, const int* in_sizes, int n_in,
                              void* d_out, int out_size, void* d_ws, size_t ws_size,
                              hipStream_t stream) {
  (void)in_sizes; (void)n_in; (void)out_size; (void)ws_size;
  const float* v     = (const float*)d_in[0];
  const float* Wq_w  = (const float*)d_in[1];
  const float* Wq_b  = (const float*)d_in[2];
  const float* Wk_w  = (const float*)d_in[3];
  const float* Wk_b  = (const float*)d_in[4];
  const float* Wv_w  = (const float*)d_in[5];
  const float* Wv_b  = (const float*)d_in[6];
  const float* lnq_g = (const float*)d_in[7];
  const float* lnq_b = (const float*)d_in[8];
  const float* lnk_g = (const float*)d_in[9];
  const float* lnk_b = (const float*)d_in[10];
  const float* Wo_w  = (const float*)d_in[11];
  const float* Wo_b  = (const float*)d_in[12];
  float* outp = (float*)d_out;

  char* ws = (char*)d_ws;
  float*    part = (float*)ws;                              // B*H*8*4096*4 = 4 MB
  _Float16* WqT  = (_Float16*)(ws + 4194304);               // 64 KB
  _Float16* WkT  = (_Float16*)(ws + 4194304 + 65536);       // 64 KB
  _Float16* WvT  = (_Float16*)(ws + 4194304 + 131072);      // 64 KB
  _Float16* WoT  = (_Float16*)(ws + 4194304 + 196608);      // 64 KB
  _Float16* MT   = (_Float16*)(ws + 4194304 + 262144);      // 256 KB

  prep_kernel<<<128, 256, 0, stream>>>(Wq_w, Wk_w, Wv_w, Wo_w, WqT, WkT, WvT, WoT);
  kv_kernel<<<B_ * H_ * 8, 128, 0, stream>>>(v, WkT, WvT, Wk_b, Wv_b, lnk_g, lnk_b, part);
  mh_kernel<<<B_ * H_, 32, 0, stream>>>(part, WoT, MT);
  out_kernel<<<B_ * 128, 128, 0, stream>>>(v, WqT, Wq_b, lnq_g, lnq_b, MT, Wo_b, outp);
}
